// TimesBlock_70549132804777
// MI455X (gfx1250) — matrix operand order repaired and checked by reading, not compiled, not run on hardware
//
#include <hip/hip_runtime.h>
#include <hip/hip_bf16.h>
#include <math.h>

// ---------------- problem constants ----------------
#define BB 64
#define TT 512
#define CC 256
#define KTOP 3
#define TMAX 768          // max folded length: T + (pv-1) <= 512+255 -> 768
#define HID 64
#define GELU_K 0.70710678118654752440f

typedef __attribute__((ext_vector_type(16))) _Float16 v16h;
typedef __attribute__((ext_vector_type(8)))  _Float16 v8h;
typedef __attribute__((ext_vector_type(8)))  float    v8f;

// ---------------- workspace layout (all sizes are 256B multiples) ------------
static const size_t SZ_BTC4  = (size_t)BB*TT*CC*4;    // 33,554,432
static const size_t SZ_BTMC2 = (size_t)BB*TMAX*CC*2;  // 25,165,824
static const size_t SZ_BTMC4 = (size_t)BB*TMAX*CC*4;  // 50,331,648
static const size_t SZ_W16   = (size_t)9*CC*CC*2;     //  1,179,648

static const size_t O_TREND  = 0;
static const size_t O_S      = O_TREND + SZ_BTC4;
static const size_t O_R      = O_S     + SZ_BTC4;
static const size_t O_IDX    = O_R     + (size_t)BB*TT*4;
static const size_t O_WPRI   = O_IDX   + (size_t)BB*KTOP*4;
static const size_t O_W16A   = O_WPRI  + (size_t)BB*KTOP*4;
static const size_t O_W16B   = O_W16A  + SZ_W16;
static const size_t O_SC1    = O_W16B  + SZ_W16;
static const size_t O_BI1    = O_SC1   + CC*4;
static const size_t O_SC2    = O_BI1   + CC*4;
static const size_t O_BI2    = O_SC2   + CC*4;
static const size_t O_Z16    = O_BI2   + CC*4;
static const size_t O_H16    = O_Z16   + SZ_BTMC2;
static const size_t O_Y      = O_H16   + SZ_BTMC2;
static const size_t O_YBAR   = O_Y     + SZ_BTMC4;
static const size_t O_SE     = O_YBAR  + (size_t)BB*CC*4;
static const size_t O_AGGNUM = O_SE    + (size_t)BB*4;
static const size_t O_AGGDEN = O_AGGNUM+ SZ_BTC4;

// ============================================================================
// 1. trend (clipped 25-tap moving average) + detrended signal s = x - trend
// ============================================================================
__global__ __launch_bounds__(256) void trend_kernel(const float* __restrict__ x,
                                                    float* __restrict__ tr,
                                                    float* __restrict__ s) {
  size_t i = (size_t)blockIdx.x*256 + threadIdx.x;
  if (i >= (size_t)BB*TT*CC) return;
  int c = (int)(i % CC);
  size_t bt = i / CC;
  int t = (int)(bt % TT);
  int b = (int)(bt / TT);
  int lo = t - 12; if (lo < 0) lo = 0;
  int hi = t + 12; if (hi > TT-1) hi = TT-1;
  const float* xb = x + ((size_t)b*TT)*CC + c;
  float sum = 0.f;
  for (int u = lo; u <= hi; ++u) sum += xb[(size_t)u*CC];
  float tv = sum / (float)(hi - lo + 1);
  tr[i] = tv;
  s[i]  = x[i] - tv;
}

// ============================================================================
// 2. circular autocorrelation: r[b,tau] = (1/C) * sum_{t,c} s[t,c]*s[(t+tau)%T,c]
//    (== irfft(mean_c |rfft(s)|^2)); only tau in [3, T/2] is needed
// ============================================================================
__global__ __launch_bounds__(256) void autocorr_kernel(const float* __restrict__ s,
                                                       float* __restrict__ r) {
  int b   = blockIdx.y;
  int tau = blockIdx.x + 3;            // 3..256
  int tid = threadIdx.x;
  const float* sb = s + (size_t)b*TT*CC;
  float acc = 0.f;
  for (int t = tid; t < TT; t += 256) {
    int t2 = t + tau; if (t2 >= TT) t2 -= TT;
    const float* p = sb + (size_t)t *CC;
    const float* q = sb + (size_t)t2*CC;
    for (int c = 0; c < CC; ++c) acc += p[c]*q[c];
  }
  __shared__ float red[256];
  red[tid] = acc; __syncthreads();
  for (int st = 128; st > 0; st >>= 1) {
    if (tid < st) red[tid] += red[tid+st];
    __syncthreads();
  }
  if (tid == 0) r[(size_t)b*TT + tau] = red[0] / (float)CC;
}

// ============================================================================
// 3. top-3 lags + softmax prior (one thread per batch row)
// ============================================================================
__global__ void topk_kernel(const float* __restrict__ r,
                            int* __restrict__ idx, float* __restrict__ wpri) {
  int b = threadIdx.x; if (b >= BB) return;
  const float* rb = r + (size_t)b*TT;
  int   sel[KTOP]; float val[KTOP];
  for (int k = 0; k < KTOP; ++k) {
    float best = -3.0e38f; int bi = 3;
    for (int tau = 3; tau <= TT/2; ++tau) {
      bool used = false;
      for (int q = 0; q < k; ++q) if (sel[q] == tau) used = true;
      float v = rb[tau];
      if (!used && v > best) { best = v; bi = tau; }
    }
    sel[k] = bi; val[k] = best;
  }
  float mx = fmaxf(val[0], fmaxf(val[1], val[2]));
  float e0 = __expf(val[0]-mx), e1 = __expf(val[1]-mx), e2 = __expf(val[2]-mx);
  float inv = 1.0f / (e0+e1+e2);
  idx [b*KTOP+0]=sel[0]; idx [b*KTOP+1]=sel[1]; idx [b*KTOP+2]=sel[2];
  wpri[b*KTOP+0]=e0*inv; wpri[b*KTOP+1]=e1*inv; wpri[b*KTOP+2]=e2*inv;
}

// ============================================================================
// 4. weight packing: OIHW f32 -> per-WMMA-fragment f16 layout.
//    Fragment = (tap, kc32, coT): 32x16 B tile stored as [lane][16 f16] with
//    lane = {khalf=lane>>4, n=lane&15}, element e <-> K = khalf*16+e
//    (this code's contiguous-k 32x16 B register image (A gathered to match) -> waves load it with 2x b128).
// ============================================================================
__global__ __launch_bounds__(256) void wpack_kernel(const float* __restrict__ w,
                                                    _Float16* __restrict__ wpk) {
  int i = blockIdx.x*256 + threadIdx.x;           // over 9*C*C
  if (i >= 9*CC*CC) return;
  int e    = i & 15;
  int lane = (i >> 4) & 31;
  int frag = i >> 9;
  int coT  = frag & 15;          // CC/16 = 16 co tiles
  int kk   = frag >> 4;
  int kc32 = kk & 7;             // CC/32 = 8 k chunks
  int tap  = kk >> 3;
  int co   = coT*16 + (lane & 15);
  int ci   = kc32*32 + (lane >> 4)*16 + e;
  wpk[i] = (_Float16)w[(size_t)co*(CC*9) + (size_t)ci*9 + tap];
}

__global__ void bnprep_kernel(const float* g1,const float* b1,const float* m1,const float* v1,
                              const float* g2,const float* b2,const float* m2,const float* v2,
                              float* sc1,float* bi1,float* sc2,float* bi2) {
  int c = threadIdx.x;  // 256
  float s1 = g1[c]*rsqrtf(v1[c]+1e-5f);
  sc1[c] = s1; bi1[c] = b1[c] - m1[c]*s1;
  float s2 = g2[c]*rsqrtf(v2[c]+1e-5f);
  sc2[c] = s2; bi2[c] = b2[c] - m2[c]*s2;
}

// ============================================================================
// 5. fold s into the period-pv 2D view, reflect-padded to Tp, convert to f16
// ============================================================================
__global__ __launch_bounds__(256) void fold_kernel(const float* __restrict__ s,
                                                   const int* __restrict__ idx, int j,
                                                   _Float16* __restrict__ z16) {
  int b = blockIdx.y;
  size_t off = (size_t)blockIdx.x*256 + threadIdx.x;
  if (off >= (size_t)TMAX*CC) return;
  int pv  = idx[b*KTOP + j];
  int pad = (pv - (TT % pv)) % pv;
  int Tp  = TT + pad;
  int m = (int)(off / CC), c = (int)(off % CC);
  if (m >= Tp) return;
  int ms = (m < TT) ? m : (2*TT - 2 - m);          // jnp.pad mode='reflect'
  z16[(size_t)b*TMAX*CC + off] = (_Float16)s[((size_t)b*TT + ms)*CC + c];
}

// ============================================================================
// 6. implicit-GEMM 3x3 SAME conv on the folded (cyc x pv) image via WMMA.
//    Block: 128 threads = 4 waves, tile M=128 x N=64; each wave owns a 32x64
//    strip (2 A fragments x 4 B fragments = 8 WMMA per k-step, doubling
//    MAC/byte on the weight stream vs a 16-row strip). A staged wave-locally
//    in LDS (no barriers: each lane stages one full 32-f16 row of its own
//    wave's strip -> in-order DS within one wave). B fragments are read
//    straight from the pre-packed register-image in global/L2 (2x b128 each).
//    K-loop: 9 taps x 8 chunks of 32.
//    mode 0: out = gelu(bn1(conv1(z)))   -> f16
//    mode 1: out = bn2(conv2(h)) + z     -> f32 (residual rebuilt from s, f32)
// ============================================================================
__global__ __launch_bounds__(128) void conv_wmma_kernel(
    const _Float16* __restrict__ Asrc,        // [B][TMAX][C] f16 (z16 or h16)
    const _Float16* __restrict__ Wpk,         // packed fragments
    const float* __restrict__ scale, const float* __restrict__ bias,
    const int* __restrict__ idx, int j,
    const float* __restrict__ s,              // residual source (mode 1)
    _Float16* __restrict__ outH, float* __restrict__ outY, int mode)
{
  // per-wave A staging: 32 rows x 32 k, row stride 48 f16 (96B, 16B-aligned)
  __shared__ alignas(16) _Float16 ldsA[4][32][48];

  const int b  = blockIdx.z;
  const int mt = blockIdx.y;                  // 0..5   (M tile of 128)
  const int nt = blockIdx.x;                  // 0..3   (N tile of 64)
  const int pv  = idx[b*KTOP + j];
  const int pad = (pv - (TT % pv)) % pv;
  const int Tp  = TT + pad;
  const int cyc = Tp / pv;
  const int m0  = mt*128;
  if (m0 >= Tp) return;                        // uniform per block: EXEC stays full

  const int tid   = threadIdx.x;
  const int wave  = tid >> 5;
  const int lane  = tid & 31;
  const int row16 = lane & 15;
  const int khalf = lane >> 4;

  // A staging: lane stages row `lane` of this wave's 32-row strip
  const int am   = m0 + wave*32 + lane;
  const int ah   = am / pv;
  const int aw   = am % pv;
  const bool amv = (am < Tp);

  const _Float16* Abase = Asrc + (size_t)b*TMAX*CC;
  const int co0 = nt*64;

  v8f acc[2][4];
  #pragma unroll
  for (int f = 0; f < 2; ++f)
    #pragma unroll
    for (int n4 = 0; n4 < 4; ++n4)
      #pragma unroll
      for (int i = 0; i < 8; ++i) acc[f][n4][i] = 0.f;

  for (int tap = 0; tap < 9; ++tap) {
    const int dh = tap/3 - 1, dw = tap%3 - 1;
    const int hh = ah + dh,   ww = aw + dw;
    const bool aok = amv && (hh >= 0) && (hh < cyc) && (ww >= 0) && (ww < pv);
    const _Float16* arow = Abase + (size_t)(hh*pv + ww)*CC;

    #pragma unroll 1
    for (int kc = 0; kc < CC; kc += 32) {
      // ---- stage A (wave-local, one full row per lane, 4x b128) ----
      v8h a0, a1, a2, a3;
      if (aok) {
        const v8h* src = (const v8h*)(arow + kc);
        a0 = src[0]; a1 = src[1]; a2 = src[2]; a3 = src[3];
      } else {
        #pragma unroll
        for (int e = 0; e < 8; ++e) {
          a0[e] = (_Float16)0.0f; a1[e] = (_Float16)0.0f;
          a2[e] = (_Float16)0.0f; a3[e] = (_Float16)0.0f;
        }
      }
      {
        v8h* dst = (v8h*)&ldsA[wave][lane][0];
        dst[0] = a0; dst[1] = a1; dst[2] = a2; dst[3] = a3;
      }

      // ---- gather 2 A fragments (contiguous k = khalf*16 + e, matching B) ----
      v16h afrag[2];
      #pragma unroll
      for (int f = 0; f < 2; ++f) {
        const int r = f*16 + row16;
        v8h lo = *(const v8h*)&ldsA[wave][r][khalf*16];
        v8h hi = *(const v8h*)&ldsA[wave][r][khalf*16 + 8];
        afrag[f] = __builtin_shufflevector(lo, hi,
                     0,1,2,3,4,5,6,7,8,9,10,11,12,13,14,15);
      }

      // ---- 4 B fragments from packed image, 8x WMMA ----
      const size_t fb = (((size_t)(tap*8 + (kc >> 5))*16) + (size_t)(nt*4))*512
                        + (size_t)lane*16;
      #pragma unroll
      for (int n4 = 0; n4 < 4; ++n4) {
        const v8h* wp = (const v8h*)(Wpk + fb + (size_t)n4*512);
        v8h b0 = wp[0], b1 = wp[1];
        v16h bmat = __builtin_shufflevector(b0, b1,
                      0,1,2,3,4,5,6,7,8,9,10,11,12,13,14,15);
        acc[0][n4] = __builtin_amdgcn_wmma_f32_16x16x32_f16(
            false, afrag[0], false, bmat, (short)0, acc[0][n4], false, false);
        acc[1][n4] = __builtin_amdgcn_wmma_f32_16x16x32_f16(
            false, afrag[1], false, bmat, (short)0, acc[1][n4], false, false);
      }
    }
  }

  // ---- epilogue: C/D layout — VGPR i -> M = (lane>>4)*8 + i, N = lane&15 ----
  #pragma unroll
  for (int f = 0; f < 2; ++f) {
    #pragma unroll
    for (int n4 = 0; n4 < 4; ++n4) {
      const int co = co0 + n4*16 + row16;
      const float sc = scale[co], bi = bias[co];
      #pragma unroll
      for (int i = 0; i < 8; ++i) {
        const int m = m0 + wave*32 + f*16 + khalf*8 + i;
        if (m < Tp) {
          float v = acc[f][n4][i]*sc + bi;
          if (mode == 0) {
            float g = 0.5f*v*(1.0f + erff(v*GELU_K));        // exact gelu
            outH[(size_t)b*TMAX*CC + (size_t)m*CC + co] = (_Float16)g;
          } else {
            int ms = (m < TT) ? m : (2*TT - 2 - m);
            float res = s[((size_t)b*TT + ms)*CC + co];
            outY[(size_t)b*TMAX*CC + (size_t)m*CC + co] = v + res;
          }
        }
      }
    }
  }
}

// ============================================================================
// 7. time-mean of y[:, :T, :], gate MLP, aggregation
// ============================================================================
__global__ __launch_bounds__(256) void mean_kernel(const float* __restrict__ y,
                                                   float* __restrict__ ybar) {
  int b = blockIdx.x, c = threadIdx.x;
  const float* yb = y + (size_t)b*TMAX*CC + c;
  float acc = 0.f;
  for (int m = 0; m < TT; ++m) acc += yb[(size_t)m*CC];
  ybar[b*CC + c] = acc / (float)TT;
}

__global__ void gate_kernel(const float* __restrict__ ybar,
                            const float* __restrict__ gw1, const float* __restrict__ gb1,
                            const float* __restrict__ gw2, const float* __restrict__ gb2,
                            const float* __restrict__ wpri, int j,
                            float* __restrict__ se_out, float* __restrict__ aggden) {
  __shared__ float hsh[HID];
  int b = blockIdx.x, tid = threadIdx.x;       // 64 threads
  const float* yb = ybar + (size_t)b*CC;
  float acc = gb1[tid];
  for (int c = 0; c < CC; ++c) acc += yb[c]*gw1[c*HID + tid];
  float g = 0.5f*acc*(1.0f + erff(acc*GELU_K));
  hsh[tid] = g*gw2[tid];
  __syncthreads();
  if (tid == 0) {
    float sum = gb2[0];
    for (int k = 0; k < HID; ++k) sum += hsh[k];
    float gate = 1.0f/(1.0f + __expf(-sum));
    // exp(log(w+eps)+log(gate+eps)) == (w+eps)*(gate+eps)
    float se = (wpri[b*KTOP + j] + 1e-8f)*(gate + 1e-8f);
    se_out[b] = se;
    aggden[b] += se;
  }
}

__global__ __launch_bounds__(256) void agg_kernel(const float* __restrict__ y,
                                                  const float* __restrict__ se,
                                                  float* __restrict__ aggnum) {
  size_t i = (size_t)blockIdx.x*256 + threadIdx.x;
  if (i >= (size_t)BB*TT*CC) return;
  size_t b   = i / ((size_t)TT*CC);
  size_t btc = i % ((size_t)TT*CC);
  aggnum[i] += y[b*(size_t)TMAX*CC + btc] * se[b];
}

__global__ __launch_bounds__(256) void init_kernel(float* aggnum, float* aggden) {
  size_t i = (size_t)blockIdx.x*256 + threadIdx.x;
  if (i < (size_t)BB*TT*CC) aggnum[i] = 0.f;
  if (i < BB)               aggden[i] = 0.f;
}

// out = agg_num/(agg_den+eps) + trend + (x - trend) = agg + x
__global__ __launch_bounds__(256) void final_kernel(const float* __restrict__ aggnum,
                                                    const float* __restrict__ aggden,
                                                    const float* __restrict__ x,
                                                    float* __restrict__ out) {
  size_t i = (size_t)blockIdx.x*256 + threadIdx.x;
  if (i >= (size_t)BB*TT*CC) return;
  size_t b = i / ((size_t)TT*CC);
  out[i] = aggnum[i]/(aggden[b] + 1e-8f) + x[i];
}

// ============================================================================
extern "C" void kernel_launch(void* const* d_in, const int* in_sizes, int n_in,
                              void* d_out, int out_size, void* d_ws, size_t ws_size,
                              hipStream_t stream) {
  const float* x    = (const float*)d_in[0];
  const float* c1w  = (const float*)d_in[1];
  const float* c2w  = (const float*)d_in[2];
  const float* g1   = (const float*)d_in[3];
  const float* be1  = (const float*)d_in[4];
  const float* m1   = (const float*)d_in[5];
  const float* v1   = (const float*)d_in[6];
  const float* g2   = (const float*)d_in[7];
  const float* be2  = (const float*)d_in[8];
  const float* m2   = (const float*)d_in[9];
  const float* v2   = (const float*)d_in[10];
  const float* gw1  = (const float*)d_in[11];
  const float* gb1  = (const float*)d_in[12];
  const float* gw2  = (const float*)d_in[13];
  const float* gb2  = (const float*)d_in[14];
  float* out = (float*)d_out;

  char* ws = (char*)d_ws;
  float*    w_tr   = (float*)   (ws + O_TREND);
  float*    w_s    = (float*)   (ws + O_S);
  float*    w_r    = (float*)   (ws + O_R);
  int*      w_idx  = (int*)     (ws + O_IDX);
  float*    w_wpri = (float*)   (ws + O_WPRI);
  _Float16* w_w16a = (_Float16*)(ws + O_W16A);
  _Float16* w_w16b = (_Float16*)(ws + O_W16B);
  float*    w_sc1  = (float*)   (ws + O_SC1);
  float*    w_bi1  = (float*)   (ws + O_BI1);
  float*    w_sc2  = (float*)   (ws + O_SC2);
  float*    w_bi2  = (float*)   (ws + O_BI2);
  _Float16* w_z16  = (_Float16*)(ws + O_Z16);
  _Float16* w_h16  = (_Float16*)(ws + O_H16);
  float*    w_y    = (float*)   (ws + O_Y);
  float*    w_ybar = (float*)   (ws + O_YBAR);
  float*    w_se   = (float*)   (ws + O_SE);
  float*    w_anum = (float*)   (ws + O_AGGNUM);
  float*    w_aden = (float*)   (ws + O_AGGDEN);

  const int NBTC = (int)(((size_t)BB*TT*CC + 255)/256);

  trend_kernel   <<<NBTC, 256, 0, stream>>>(x, w_tr, w_s);
  autocorr_kernel<<<dim3(TT/2 - 2, BB), 256, 0, stream>>>(w_s, w_r);
  topk_kernel    <<<1, 64, 0, stream>>>(w_r, w_idx, w_wpri);
  bnprep_kernel  <<<1, 256, 0, stream>>>(g1, be1, m1, v1, g2, be2, m2, v2,
                                         w_sc1, w_bi1, w_sc2, w_bi2);
  wpack_kernel   <<<(9*CC*CC)/256, 256, 0, stream>>>(c1w, w_w16a);
  wpack_kernel   <<<(9*CC*CC)/256, 256, 0, stream>>>(c2w, w_w16b);
  init_kernel    <<<NBTC, 256, 0, stream>>>(w_anum, w_aden);

  for (int j = 0; j < KTOP; ++j) {
    fold_kernel<<<dim3((TMAX*CC)/256, BB), 256, 0, stream>>>(w_s, w_idx, j, w_z16);
    // conv1 + bn1 + gelu  (f16 out)
    conv_wmma_kernel<<<dim3(CC/64, TMAX/128, BB), 128, 0, stream>>>(
        w_z16, w_w16a, w_sc1, w_bi1, w_idx, j, w_s, w_h16, (float*)nullptr, 0);
    // conv2 + bn2 + residual (f32 out)
    conv_wmma_kernel<<<dim3(CC/64, TMAX/128, BB), 128, 0, stream>>>(
        w_h16, w_w16b, w_sc2, w_bi2, w_idx, j, w_s, (_Float16*)nullptr, w_y, 1);
    mean_kernel<<<BB, 256, 0, stream>>>(w_y, w_ybar);
    gate_kernel<<<BB, HID, 0, stream>>>(w_ybar, gw1, gb1, gw2, gb2,
                                        w_wpri, j, w_se, w_aden);
    agg_kernel <<<NBTC, 256, 0, stream>>>(w_y, w_se, w_anum);
  }
  final_kernel<<<NBTC, 256, 0, stream>>>(w_anum, w_aden, x, out);
}
